// WFSA_88270167867759
// MI455X (gfx1250) — matrix operand order repaired and checked by reading, not compiled, not run on hardware
//
#include <hip/hip_runtime.h>
#include <cstdint>

// WFSA forward pass re-associated as a chunked matrix-product scan.
//   x_T^T = init^T * A_{u0} * A_{u1} * ... * A_{u4095}
//
// Phase 0: one-time split-transpose of A: fp32 [q][u][q'] ->
//          bf16 hi/lo planes [u][q'][q] in workspace (k-contiguous, so WMMA
//          B-fragments are two contiguous 16B runs loadable by b128).
// Phase 1: 64 chunks x 64 tokens; chunk products via split-bf16 WMMA
//          (hi*hi + hi*lo + lo*hi, fp32 acc). Running 64-row strip kept as
//          hi/lo bf16 planes in LDS; one WG per strip, 512 WGs.
// Phase 2: x <- P_c^T x over 64 chunk matrices (stream-ordered GEMVs),
//          then out = <x, final>.
//
// Workspace: 512MB AhiT + 512MB AloT + 64MB P + x buffers  (~1.07 GiB).

#define QN 512
#define SN 1024
#define TN 4096
#define CHUNK 64
#define NCHUNK (TN / CHUNK)      // 64
#define STRIP 64
#define NSTRIP (QN / STRIP)      // 8
#define THREADS 256              // 8 waves (wave32)
#define SQ ((size_t)SN * QN)     // q-row stride of A
#define MROW (QN + 16)           // padded M row (520 bf16 = 260 words, %64 = 4)

typedef __attribute__((ext_vector_type(16))) __bf16 v16bf;
typedef __attribute__((ext_vector_type(8)))  __bf16 v8bf;
typedef __attribute__((ext_vector_type(8)))  float  v8f;

// Concatenate two contiguous 16B runs into a WMMA operand (no element moves).
__device__ __forceinline__ v16bf frag2(const __bf16* p0, const __bf16* p1) {
    v8bf a = *(const v8bf*)p0;
    v8bf b = *(const v8bf*)p1;
    return __builtin_shufflevector(a, b, 0, 1, 2, 3, 4, 5, 6, 7,
                                         8, 9, 10, 11, 12, 13, 14, 15);
}

// ---------------------------------------------------------------------------
// Phase 0: split-transpose  A[q][u][c] (f32)  ->  AhiT/AloT[u][c][q] (bf16).
// One block per (u, 64-column tile); LDS bounce for the transpose.
// ---------------------------------------------------------------------------
#define CT 64
#define TP (CT + 2)
__global__ __launch_bounds__(256)
void wfsa_split_transpose(const float* __restrict__ A,
                          __bf16* __restrict__ AhiT,
                          __bf16* __restrict__ AloT)
{
    extern __shared__ unsigned char smem[];
    float* tile = (float*)smem;                    // [QN][TP]
    const int u   = blockIdx.x >> 3;
    const int ct0 = (blockIdx.x & 7) * CT;
    const int tid = threadIdx.x;

    for (int idx = tid; idx < QN * (CT / 4); idx += 256) {
        const int q  = idx >> 4;                   // CT/4 = 16 float4 per row
        const int c4 = idx & 15;
        const float4 v = *(const float4*)(A + (size_t)q * SQ + (size_t)u * QN
                                            + ct0 + c4 * 4);
        float* d = tile + q * TP + c4 * 4;
        d[0] = v.x; d[1] = v.y; d[2] = v.z; d[3] = v.w;
    }
    __syncthreads();

    for (int idx = tid; idx < CT * (QN / 8); idx += 256) {
        const int c  = idx >> 6;                   // QN/8 = 64 q-runs per col
        const int q0 = (idx & 63) * 8;
        v8bf h, l;
#pragma unroll
        for (int j = 0; j < 8; ++j) {
            const float f = tile[(q0 + j) * TP + c];
            const __bf16 hb = (__bf16)f;
            h[j] = hb;
            l[j] = (__bf16)(f - (float)hb);
        }
        const size_t off = ((size_t)u * QN + ct0 + c) * QN + q0;
        *(v8bf*)(AhiT + off) = h;
        *(v8bf*)(AloT + off) = l;
    }
}

// ---------------------------------------------------------------------------
// Phase 1: per-chunk running product, one 64-row strip per workgroup.
// LDS: Mhi/Mlo bf16 [STRIP][MROW] (~130 KB). B operands stream from the
// pre-split transposed planes in global memory (k-contiguous b128 loads).
// ---------------------------------------------------------------------------
__global__ __launch_bounds__(THREADS)
void wfsa_chunk_products(const float* __restrict__ A,
                         const __bf16* __restrict__ AhiT,
                         const __bf16* __restrict__ AloT,
                         const int* __restrict__ tokens,
                         float* __restrict__ P)    // [NCHUNK][QN][QN]
{
    extern __shared__ unsigned char smem[];
    __bf16* Mhi = (__bf16*)smem;                   // [STRIP][MROW]
    __bf16* Mlo = Mhi + STRIP * MROW;

    const int chunk = blockIdx.x / NSTRIP;
    const int strip = blockIdx.x % NSTRIP;
    const int tid   = threadIdx.x;
    const int lane  = tid & 31;
    const int wave  = tid >> 5;
    const int t0    = chunk * CHUNK;

    // ---- init M = rows [strip*64, +64) of A_{u(t0)}, split into hi/lo ------
    {
        const int u0 = tokens[t0];
        for (int idx = tid; idx < STRIP * (QN / 4); idx += THREADS) {
            const int row = idx >> 7;
            const int c4  = idx & 127;
            const float4 v = *(const float4*)(
                A + (size_t)(strip * STRIP + row) * SQ + (size_t)u0 * QN + c4 * 4);
            const __bf16 h0 = (__bf16)v.x, h1 = (__bf16)v.y,
                         h2 = (__bf16)v.z, h3 = (__bf16)v.w;
            __bf16* dh = Mhi + row * MROW + c4 * 4;
            __bf16* dl = Mlo + row * MROW + c4 * 4;
            dh[0] = h0; dh[1] = h1; dh[2] = h2; dh[3] = h3;
            dl[0] = (__bf16)(v.x - (float)h0);
            dl[1] = (__bf16)(v.y - (float)h1);
            dl[2] = (__bf16)(v.z - (float)h2);
            dl[3] = (__bf16)(v.w - (float)h3);
        }
    }
    __syncthreads();

    // Wave owns 4 row-tiles x 4 col-tiles; every B fragment loaded once per WG.
    const int rlo    = lane & 15;
    const int khalfA = (lane >> 4) * 8;   // A-frag K interleave (ISA 7.12.2)
    const int kbaseB = (lane >> 4) * 8;   // B-frag K interleave (matches A)
    const int ct0w   = wave * 4;          // this wave's first col-tile

    for (int s = 1; s < CHUNK; ++s) {
        const int u = tokens[t0 + s];
        const __bf16* __restrict__ Bh = AhiT + (size_t)u * QN * QN; // [c][k]
        const __bf16* __restrict__ Bl = AloT + (size_t)u * QN * QN;

        // Warm a slice of the next token's hi plane.
        const int un = (s + 1 < CHUNK) ? tokens[t0 + s + 1] : u;
        __builtin_prefetch(AhiT + (size_t)un * QN * QN + (size_t)tid * 1024, 0, 1);

        v8f acc[16];
        const v8f zero = {0.f, 0.f, 0.f, 0.f, 0.f, 0.f, 0.f, 0.f};
#pragma unroll
        for (int i = 0; i < 16; ++i) acc[i] = zero;

        for (int kb = 0; kb < QN; kb += 32) {
            v16bf Ah[4], Al[4], Bhf[4], Blf[4];
#pragma unroll
            for (int rt = 0; rt < 4; ++rt) {
                const int base = (rt * 16 + rlo) * MROW + kb + khalfA;
                Ah[rt] = frag2(Mhi + base, Mhi + base + 16);
                Al[rt] = frag2(Mlo + base, Mlo + base + 16);
            }
#pragma unroll
            for (int cj = 0; cj < 4; ++cj) {
                const size_t base = (size_t)((ct0w + cj) * 16 + rlo) * QN
                                  + kb + kbaseB;
                Bhf[cj] = frag2(Bh + base, Bh + base + 16);
                Blf[cj] = frag2(Bl + base, Bl + base + 16);
            }
#pragma unroll
            for (int rt = 0; rt < 4; ++rt)
#pragma unroll
                for (int cj = 0; cj < 4; ++cj) {
                    const int t = rt * 4 + cj;
                    acc[t] = __builtin_amdgcn_wmma_f32_16x16x32_bf16(
                        false, Ah[rt], false, Bhf[cj], (short)0, acc[t], false, false);
                    acc[t] = __builtin_amdgcn_wmma_f32_16x16x32_bf16(
                        false, Ah[rt], false, Blf[cj], (short)0, acc[t], false, false);
                    acc[t] = __builtin_amdgcn_wmma_f32_16x16x32_bf16(
                        false, Al[rt], false, Bhf[cj], (short)0, acc[t], false, false);
                }
        }
        __syncthreads();   // all waves done reading M for this step

        // Write D tiles back as split hi/lo (C/D layout: M = v + 8*(lane>>4)).
#pragma unroll
        for (int rt = 0; rt < 4; ++rt) {
            const int rbase = rt * 16 + 8 * (lane >> 4);
#pragma unroll
            for (int cj = 0; cj < 4; ++cj) {
                const int col = (ct0w + cj) * 16 + rlo;
                const v8f a = acc[rt * 4 + cj];
#pragma unroll
                for (int v = 0; v < 8; ++v) {
                    const float f = a[v];
                    const __bf16 hb = (__bf16)f;
                    Mhi[(rbase + v) * MROW + col] = hb;
                    Mlo[(rbase + v) * MROW + col] = (__bf16)(f - (float)hb);
                }
            }
        }
        __syncthreads();
    }

    // Emit this strip of the chunk product (recombine hi + lo).
    float* __restrict__ Pc = P + (size_t)chunk * QN * QN + (size_t)strip * STRIP * QN;
    for (int idx = tid; idx < STRIP * (QN / 4); idx += THREADS) {
        const int row = idx >> 7;
        const int c4  = idx & 127;
        const __bf16* sh = Mhi + row * MROW + c4 * 4;
        const __bf16* sl = Mlo + row * MROW + c4 * 4;
        float4 o;
        o.x = (float)sh[0] + (float)sl[0];
        o.y = (float)sh[1] + (float)sl[1];
        o.z = (float)sh[2] + (float)sl[2];
        o.w = (float)sh[3] + (float)sl[3];
        *(float4*)(Pc + row * QN + c4 * 4) = o;
    }
}

// ---------------------------------------------------------------------------
// Phase 2 kernels
// ---------------------------------------------------------------------------
__global__ void wfsa_init_x(float* __restrict__ x, const float* __restrict__ init) {
    const int j = blockIdx.x * blockDim.x + threadIdx.x;
    if (j < QN) x[j] = init[j];
}

__global__ void wfsa_gemv(const float* __restrict__ P,
                          const float* __restrict__ xin,
                          float* __restrict__ xout) {
    const int j = blockIdx.x * blockDim.x + threadIdx.x;
    float s = 0.f;
#pragma unroll 8
    for (int i = 0; i < QN; ++i)
        s = fmaf(P[(size_t)i * QN + j], xin[i], s);
    xout[j] = s;
}

__global__ void wfsa_dot(const float* __restrict__ x,
                         const float* __restrict__ fin,
                         float* __restrict__ out) {
    __shared__ float red[QN];
    const int t = threadIdx.x;
    red[t] = x[t] * fin[t];
    __syncthreads();
    for (int w = QN / 2; w > 0; w >>= 1) {
        if (t < w) red[t] += red[t + w];
        __syncthreads();
    }
    if (t == 0) out[0] = red[0];
}

// ---------------------------------------------------------------------------
// Launcher
// ---------------------------------------------------------------------------
extern "C" void kernel_launch(void* const* d_in, const int* in_sizes, int n_in,
                              void* d_out, int out_size, void* d_ws, size_t ws_size,
                              hipStream_t stream) {
    const float* A      = (const float*)d_in[0];
    const float* initv  = (const float*)d_in[1];
    const float* finalv = (const float*)d_in[2];
    const int*   tokens = (const int*)d_in[3];

    const size_t PLANE = (size_t)SN * QN * QN * sizeof(__bf16);   // 512 MiB
    __bf16* AhiT = (__bf16*)d_ws;
    __bf16* AloT = (__bf16*)((char*)d_ws + PLANE);
    float*  P    = (float*)((char*)d_ws + 2 * PLANE);             // 64 MiB
    float*  xa   = (float*)((char*)P + (size_t)NCHUNK * QN * QN * sizeof(float));
    float*  xb   = xa + QN;

    const size_t smem0 = (size_t)QN * TP * sizeof(float);          // ~132 KB
    wfsa_split_transpose<<<dim3(SN * (QN / CT)), dim3(256), smem0, stream>>>(
        A, AhiT, AloT);

    const size_t smem1 = (size_t)2 * STRIP * MROW * sizeof(__bf16); // ~130 KB
    wfsa_chunk_products<<<dim3(NCHUNK * NSTRIP), dim3(THREADS), smem1, stream>>>(
        A, AhiT, AloT, tokens, P);

    wfsa_init_x<<<dim3(2), dim3(256), 0, stream>>>(xa, initv);

    float* cur = xa;
    float* nxt = xb;
    for (int c = 0; c < NCHUNK; ++c) {
        const float* Pc = P + (size_t)c * QN * QN;
        wfsa_gemv<<<dim3(8), dim3(64), 0, stream>>>(Pc, cur, nxt);
        float* t = cur; cur = nxt; nxt = t;
    }

    wfsa_dot<<<dim3(1), dim3(QN), 0, stream>>>(cur, finalv, (float*)d_out);
}
